// HybridMultiHeadAttention_45964740002182
// MI455X (gfx1250) — matrix operand order repaired and checked by reading, not compiled, not run on hardware
//
#include <hip/hip_runtime.h>
#include <hip/hip_bf16.h>

// MI455X / gfx1250, wave32. f16 WMMA (16x16x32) + f32 accum, async global->LDS staging.

typedef _Float16 v16h __attribute__((ext_vector_type(16)));
typedef _Float16 v8h  __attribute__((ext_vector_type(8)));
typedef float    v8f  __attribute__((ext_vector_type(8)));

// ---------------------------------------------------------------------------
// CDNA5 async global->LDS staging (ASYNCcnt tracked, no VGPR round-trip)
// ---------------------------------------------------------------------------
__device__ __forceinline__ void async_b128(uint32_t lds_off, uint64_t gaddr) {
  asm volatile("global_load_async_to_lds_b128 %0, %1, off"
               :: "v"(lds_off), "v"(gaddr) : "memory");
}
__device__ __forceinline__ void async_wait0() {
  asm volatile("s_wait_asynccnt 0x0" ::: "memory");
}
// stage 32 consecutive halves (64B) from global to LDS as 4 async b128 ops
__device__ __forceinline__ void issue_row32(uint32_t lds_off, const _Float16* g) {
#pragma unroll
  for (int i = 0; i < 4; ++i)
    async_b128(lds_off + 16u * i, (uint64_t)(uintptr_t)(g + 8 * i));
}
__device__ __forceinline__ uint32_t lds_off_of(const _Float16* p) {
  return (uint32_t)(uintptr_t)p;   // low 32 bits of flat addr == LDS byte offset
}

// ---------------------------------------------------------------------------
// WMMA fragment helpers (wave32 layouts per CDNA5 ISA 7.12.2)
// A 16x32 f16:  M = lane&15 ; VGPR v, half p -> K = (v>=4?16:0)+(lane>=16?8:0)+(v&3)*2+p
//   => per lane: halves [kh..kh+7] and [16+kh..16+kh+7], kh = (lane>=16)*8 : contiguous!
// B 32x16 f16:  N = lane&15 ; VGPR v, half p -> K = (v>=4?16:0)+(lane>=16?8:0)+(v&3)*2+p
//   => per lane: halves [kb..kb+7] and [16+kb..16+kb+7], kb = (lane>=16)*8 : same as A
// C/D 16x16 f32: N = lane&15 ; VGPR v -> M = (lane>=16?8:0)+v
// ---------------------------------------------------------------------------
__device__ __forceinline__ v8f wmma_f16(v16h a, v16h b, v8f c) {
  return __builtin_amdgcn_wmma_f32_16x16x32_f16(false, a, false, b, (short)0, c,
                                                false, false);
}
__device__ __forceinline__ v16h cat8(v8h lo, v8h hi) {
  return __builtin_shufflevector(lo, hi, 0, 1, 2, 3, 4, 5, 6, 7,
                                 8, 9, 10, 11, 12, 13, 14, 15);
}

// A fragment: A[m][k] = lds[(row0+m)*ld + col0 + k]  -> 2 x ds_load_b128 per lane
__device__ __forceinline__ v16h load_a(const _Float16* lds, int row0, int col0, int ld) {
  const int lane = threadIdx.x & 31;
  const int m    = lane & 15;
  const int kh   = (lane >> 4) << 3;
  const _Float16* p =
      (const _Float16*)__builtin_assume_aligned(lds + (row0 + m) * ld + col0 + kh, 4);
  v8h lo = *(const v8h*)(p);
  v8h hi = *(const v8h*)(p + 16);
  return cat8(lo, hi);
}

// B[k][n] = lds[(row0+n)*ld + col0 + k]  (source rows are the N dim)
// -> per lane: halves [kb..kb+7] and [16+kb..16+kb+7], kb = (lane>=16)*8 = 2 x ds_load_b128
__device__ __forceinline__ v16h load_bt(const _Float16* lds, int row0, int col0, int ld) {
  const int lane = threadIdx.x & 31;
  const int n    = lane & 15;
  const int kb   = (lane >> 4) << 3;
  const _Float16* p =
      (const _Float16*)__builtin_assume_aligned(lds + (row0 + n) * ld + col0 + kb, 4);
  v8h lo = *(const v8h*)(p);
  v8h hi = *(const v8h*)(p + 16);
  return cat8(lo, hi);
}

// ---------------------------------------------------------------------------
// TT core expansion:  W[o,d] = sum c0[0,m,i,r1] c1[r1,p,j,r2] c2[r2,q,k,r3] c3[r3,u,l,0]
// ---------------------------------------------------------------------------
__global__ void tt_stage1(const float* __restrict__ c0, const float* __restrict__ c1,
                          float* __restrict__ W2) {   // W2[mp(16)][ij(16)][r2(64)]
  int idx = blockIdx.x * blockDim.x + threadIdx.x;
  if (idx >= 16 * 16 * 64) return;
  int r2 = idx & 63;
  int ij = (idx >> 6) & 15;
  int mp = idx >> 10;
  int m = mp >> 2, p = mp & 3, i = ij >> 2, j = ij & 3;
  float s = 0.f;
#pragma unroll 8
  for (int r1 = 0; r1 < 32; ++r1)
    s += c0[(m * 4 + i) * 32 + r1] * c1[((r1 * 4 + p) * 4 + j) * 64 + r2];
  W2[idx] = s;
}

__global__ void tt_stage2(const float* __restrict__ W2, const float* __restrict__ c2,
                          float* __restrict__ W3) {   // W3[mpq(96)][ijk(96)][r3(32)]
  int idx = blockIdx.x * blockDim.x + threadIdx.x;
  if (idx >= 96 * 96 * 32) return;
  int r3  = idx & 31;
  int col = (idx >> 5) % 96;
  int row = (idx >> 5) / 96;
  int mp = row / 6, q = row % 6, ij = col / 6, k = col % 6;
  float s = 0.f;
#pragma unroll 8
  for (int r2 = 0; r2 < 64; ++r2)
    s += W2[(mp * 16 + ij) * 64 + r2] * c2[((r2 * 6 + q) * 6 + k) * 32 + r3];
  W3[idx] = s;
}

__global__ void tt_stage3(const float* __restrict__ W3, const float* __restrict__ c3,
                          _Float16* __restrict__ W16) {  // W16[o(768)][d(768)]
  int idx = blockIdx.x * blockDim.x + threadIdx.x;
  if (idx >= 768 * 768) return;
  int o = idx / 768, d = idx - o * 768;
  int mpq = o >> 3, u = o & 7;
  int ijk = d >> 3, l = d & 7;
  float s = 0.f;
#pragma unroll 8
  for (int r3 = 0; r3 < 32; ++r3)
    s += W3[(mpq * 96 + ijk) * 32 + r3] * c3[(r3 * 8 + u) * 8 + l];
  W16[idx] = (_Float16)s;
}

__global__ void f32_to_f16(const float* __restrict__ in, _Float16* __restrict__ out, int n) {
  int i = blockIdx.x * blockDim.x + threadIdx.x;
  if (i < n) out[i] = (_Float16)in[i];
}

// ---------------------------------------------------------------------------
// GEMM: C[n,o] = sum_d A16[n,d] * W16[o,d] + bias[o]
// 64x64 tile, 128 threads (4 waves), K-step 64, double-buffered async LDS staging.
// MODE 0: f16 per-head layout [B,H,S,64] (Q,K)
// MODE 1: f32 contiguous [4096,768]      (final output)
// MODE 2: f16 transposed per-head layout [B,H,64,S] (V, so P@V B-frags are contiguous)
// ---------------------------------------------------------------------------
#define GLD    72  // 64 + 8 halves pad
#define KITERS 12  // 768 / 64

template <int MODE>
__global__ __launch_bounds__(128) void tt_gemm(const _Float16* __restrict__ A,
                                               const _Float16* __restrict__ W,
                                               const float* __restrict__ bias,
                                               _Float16* __restrict__ out16,
                                               float* __restrict__ out32) {
  __shared__ alignas(16) _Float16 As[2][64 * GLD];
  __shared__ alignas(16) _Float16 Ws[2][64 * GLD];
  const int m0 = blockIdx.x * 64;
  const int n0 = blockIdx.y * 64;
  const int wave = threadIdx.x >> 5;
  const int lane = threadIdx.x & 31;
  const int lrow = threadIdx.x >> 1;         // 0..63
  const int lseg = (threadIdx.x & 1) * 32;   // 0 or 32 halves

  const _Float16* ga = A + (size_t)(m0 + lrow) * 768 + lseg;
  const _Float16* gw = W + (size_t)(n0 + lrow) * 768 + lseg;
  uint32_t laBuf[2], lwBuf[2];
#pragma unroll
  for (int b = 0; b < 2; ++b) {
    laBuf[b] = lds_off_of(&As[b][0] + lrow * GLD + lseg);
    lwBuf[b] = lds_off_of(&Ws[b][0] + lrow * GLD + lseg);
  }

  // prologue: stage tile 0 into buffer 0
  issue_row32(laBuf[0], ga);
  issue_row32(lwBuf[0], gw);
  async_wait0();
  __syncthreads();

  v8f acc[4] = {};
  for (int it = 0; it < KITERS; ++it) {
    const int cur = it & 1;
    if (it + 1 < KITERS) {  // prefetch next tile into the other buffer
      issue_row32(laBuf[cur ^ 1], ga + (it + 1) * 64);
      issue_row32(lwBuf[cur ^ 1], gw + (it + 1) * 64);
    }
    const _Float16* as = As[cur];
    const _Float16* ws = Ws[cur];
    v16h a0 = load_a(as, wave * 16, 0, GLD);
    v16h a1 = load_a(as, wave * 16, 32, GLD);
#pragma unroll
    for (int nt = 0; nt < 4; ++nt) {
      acc[nt] = wmma_f16(a0, load_bt(ws, nt * 16, 0, GLD), acc[nt]);
      acc[nt] = wmma_f16(a1, load_bt(ws, nt * 16, 32, GLD), acc[nt]);
    }
    async_wait0();
    __syncthreads();
  }

  const int mbase = m0 + wave * 16 + ((lane >> 4) << 3);
  const int nbase = n0 + (lane & 15);
#pragma unroll
  for (int nt = 0; nt < 4; ++nt) {
    int col = nbase + nt * 16;
    float bv = bias[col];
#pragma unroll
    for (int v = 0; v < 8; ++v) {
      int row = mbase + v;
      float val = acc[nt][v] + bv;
      int bb = row >> 11, s = row & 2047;        // B=2, S=2048
      int h = col >> 6, dd = col & 63;           // H=12, Dh=64
      if constexpr (MODE == 0) {
        out16[(((size_t)bb * 12 + h) * 2048 + s) * 64 + dd] = (_Float16)val;
      } else if constexpr (MODE == 2) {
        out16[(((size_t)bb * 12 + h) * 64 + dd) * 2048 + s] = (_Float16)val;
      } else {
        out32[(size_t)row * 768 + col] = val;
      }
    }
  }
}

// ---------------------------------------------------------------------------
// Causal flash attention per (b, h, 64-row query tile). 128 threads (4 waves).
// QK^T and P@V via WMMA; online softmax in LDS (f32); async K/V staging.
// V arrives transposed per head ([B,H,64,S]) so all B-fragments are contiguous.
// ---------------------------------------------------------------------------
#define ALD  72  // 64 + 8 halves pad
#define ALDS 68  // 64 + 4 f32 pad

__global__ __launch_bounds__(128) void tt_attn(const _Float16* __restrict__ Q,
                                               const _Float16* __restrict__ K,
                                               const _Float16* __restrict__ Vt,
                                               _Float16* __restrict__ attn_out) {
  __shared__ alignas(16) _Float16 Qs[64 * ALD];
  __shared__ alignas(16) _Float16 Ks[64 * ALD];
  __shared__ alignas(16) _Float16 Vs[64 * ALD];  // [d][key] tile
  __shared__ alignas(16) float    Ss[64 * ALDS];
  __shared__ alignas(16) _Float16 Ps[64 * ALD];
  __shared__ float rowf[64];

  const int b = blockIdx.z, h = blockIdx.y, qt = blockIdx.x;
  const int q0 = qt * 64;
  const size_t head = ((size_t)b * 12 + h) * 2048 * 64;
  const _Float16* Qh  = Q + head;
  const _Float16* Kh  = K + head;
  const _Float16* Vth = Vt + head;  // [64 d][2048 s]

  const int wave = threadIdx.x >> 5;
  const int lane = threadIdx.x & 31;
  const int lrow = threadIdx.x >> 1;
  const int lseg = (threadIdx.x & 1) * 32;

  const uint32_t lq = lds_off_of(Qs + lrow * ALD + lseg);
  const uint32_t lk = lds_off_of(Ks + lrow * ALD + lseg);
  const uint32_t lv = lds_off_of(Vs + lrow * ALD + lseg);

  // stage Q tile once (async)
  issue_row32(lq, Qh + (size_t)(q0 + lrow) * 64 + lseg);

  v8f acc[4] = {};
  float mrun = -3.0e38f, lrun = 0.0f;
  const int Mb = wave * 16 + ((lane >> 4) << 3);

  for (int kb = 0; kb <= qt; ++kb) {
    const int k0 = kb * 64;
    issue_row32(lk, Kh + (size_t)(k0 + lrow) * 64 + lseg);
    issue_row32(lv, Vth + (size_t)lrow * 2048 + k0 + lseg);  // Vs[d][key]
    async_wait0();
    __syncthreads();

    // S = (Q K^T) * scale, causal mask on diagonal tile
    v16h aq0 = load_a(Qs, wave * 16, 0, ALD);
    v16h aq1 = load_a(Qs, wave * 16, 32, ALD);
#pragma unroll
    for (int nt = 0; nt < 4; ++nt) {
      v8f s = {};
      s = wmma_f16(aq0, load_bt(Ks, nt * 16, 0, ALD), s);
      s = wmma_f16(aq1, load_bt(Ks, nt * 16, 32, ALD), s);
      const int N = nt * 16 + (lane & 15);
#pragma unroll
      for (int v = 0; v < 8; ++v) {
        float sv = s[v] * 0.125f;  // 1/sqrt(64)
        if (k0 + N > q0 + Mb + v) sv = -10000.0f;
        Ss[(Mb + v) * ALDS + N] = sv;
      }
    }
    __syncthreads();

    // online softmax: one thread per query row
    if (threadIdx.x < 64) {
      const float* srow = Ss + threadIdx.x * ALDS;
      float rmax = srow[0];
#pragma unroll 8
      for (int c = 1; c < 64; ++c) rmax = fmaxf(rmax, srow[c]);
      float mnew = fmaxf(mrun, rmax);
      float fac  = __expf(mrun - mnew);
      float psum = 0.0f;
      _Float16* prow = Ps + threadIdx.x * ALD;
#pragma unroll 8
      for (int c = 0; c < 64; ++c) {
        float pv = __expf(srow[c] - mnew);
        prow[c]  = (_Float16)pv;
        psum += pv;
      }
      lrun = lrun * fac + psum;
      mrun = mnew;
      rowf[threadIdx.x] = fac;
    }
    __syncthreads();

    // rescale running accumulator, then acc += P @ V
    {
      float f[8];
#pragma unroll
      for (int v = 0; v < 8; ++v) f[v] = rowf[Mb + v];
#pragma unroll
      for (int nt = 0; nt < 4; ++nt)
#pragma unroll
        for (int v = 0; v < 8; ++v) acc[nt][v] *= f[v];
      v16h ap0 = load_a(Ps, wave * 16, 0, ALD);
      v16h ap1 = load_a(Ps, wave * 16, 32, ALD);
      // B[kk][n=d] = V[k0+kk][d] = Vs[d][kk]  -> contiguous load_bt on Vs
#pragma unroll
      for (int nt = 0; nt < 4; ++nt) {
        acc[nt] = wmma_f16(ap0, load_bt(Vs, nt * 16, 0, ALD), acc[nt]);
        acc[nt] = wmma_f16(ap1, load_bt(Vs, nt * 16, 32, ALD), acc[nt]);
      }
    }
    __syncthreads();
  }

  if (threadIdx.x < 64) rowf[threadIdx.x] = 1.0f / lrun;
  __syncthreads();
  {
    const int n = lane & 15;
#pragma unroll
    for (int v = 0; v < 8; ++v) {
      float inv = rowf[Mb + v];
      size_t grow = (size_t)b * 2048 + q0 + Mb + v;
#pragma unroll
      for (int nt = 0; nt < 4; ++nt) {
        attn_out[grow * 768 + h * 64 + nt * 16 + n] = (_Float16)(acc[nt][v] * inv);
      }
    }
  }
}

// ---------------------------------------------------------------------------
extern "C" void kernel_launch(void* const* d_in, const int* in_sizes, int n_in,
                              void* d_out, int out_size, void* d_ws, size_t ws_size,
                              hipStream_t stream) {
  const float* x = (const float*)d_in[0];
  // d_in[1] = causal mask (ignored; computed analytically)
  const float* cores[4][4];
  const float* biases[4];
  for (int p = 0; p < 4; ++p) {
    for (int c = 0; c < 4; ++c) cores[p][c] = (const float*)d_in[2 + p * 5 + c];
    biases[p] = (const float*)d_in[2 + p * 5 + 4];
  }

  char* ws = (char*)d_ws;
  size_t off = 0;
  auto alloc = [&](size_t bytes) -> void* {
    void* p = ws + off;
    off += (bytes + 255) & ~(size_t)255;
    return p;
  };
  const size_t NTOK = 4096;  // B*S
  _Float16* X16 = (_Float16*)alloc(NTOK * 768 * 2);
  _Float16* W16[4];
  for (int p = 0; p < 4; ++p) W16[p] = (_Float16*)alloc(768ull * 768 * 2);
  _Float16* Q16    = (_Float16*)alloc(NTOK * 768 * 2);
  _Float16* K16    = (_Float16*)alloc(NTOK * 768 * 2);
  _Float16* V16t   = (_Float16*)alloc(NTOK * 768 * 2);
  _Float16* attn16 = (_Float16*)alloc(NTOK * 768 * 2);
  float* W2s = (float*)alloc(16 * 16 * 64 * 4);
  float* W3s = (float*)alloc(96ull * 96 * 32 * 4);

  // 1) x -> f16
  f32_to_f16<<<(int)((NTOK * 768 + 255) / 256), 256, 0, stream>>>(x, X16,
                                                                  (int)(NTOK * 768));
  // 2) expand TT cores to dense 768x768 f16 weights (stream-serialized scratch reuse)
  for (int p = 0; p < 4; ++p) {
    tt_stage1<<<64, 256, 0, stream>>>(cores[p][0], cores[p][1], W2s);
    tt_stage2<<<1152, 256, 0, stream>>>(W2s, cores[p][2], W3s);
    tt_stage3<<<2304, 256, 0, stream>>>(W3s, cores[p][3], W16[p]);
  }
  // 3) Q/K/V projections (WMMA GEMM); V written transposed per head
  dim3 ggrid(64, 12);
  tt_gemm<0><<<ggrid, 128, 0, stream>>>(X16, W16[0], biases[0], Q16, nullptr);
  tt_gemm<0><<<ggrid, 128, 0, stream>>>(X16, W16[1], biases[1], K16, nullptr);
  tt_gemm<2><<<ggrid, 128, 0, stream>>>(X16, W16[2], biases[2], V16t, nullptr);
  // 4) causal flash attention
  tt_attn<<<dim3(32, 12, 2), 128, 0, stream>>>(Q16, K16, V16t, attn16);
  // 5) output projection -> f32 d_out
  tt_gemm<1><<<ggrid, 128, 0, stream>>>(attn16, W16[3], biases[3], nullptr,
                                        (float*)d_out);
}
